// RvNN_8091718386192
// MI455X (gfx1250) — matrix operand order repaired and checked by reading, not compiled, not run on hardware
//
#include <hip/hip_runtime.h>
#include <hip/hip_bf16.h>
#include <stdint.h>

// ---------------- problem constants (from reference) ----------------
#define NN 4096   // nodes
#define DD 5000   // input feature dim
#define HH 512    // hidden dim
#define CC 4      // classes
#define LL 8      // tree levels

typedef __bf16 bf16_t;
typedef __attribute__((ext_vector_type(8)))  float   v8f;
typedef __attribute__((ext_vector_type(8)))  __bf16  v8bf;
typedef __attribute__((ext_vector_type(16))) __bf16  v16bf;

// ---------------- WMMA fragment loaders (CDNA5 16-bit layouts) ----------------
// A 16x32 (MxK): lanes 0-15 hold row M=lane, K = {0..7, 16..23}; lanes 16-31 hold
// row M=lane-16, K = {8..15, 24..31}.  Two contiguous 8-elem (16B) chunks.
__device__ __forceinline__ v16bf cat16(v8bf a, v8bf b) {
    return __builtin_shufflevector(a, b, 0,1,2,3,4,5,6,7,8,9,10,11,12,13,14,15);
}

__device__ __forceinline__ v16bf load_a_frag(const bf16_t* __restrict__ A, int lda,
                                             int row, int k0, int lane) {
    const bf16_t* p = A + (size_t)row * lda + k0 + ((lane >> 4) << 3);
    v8bf c0 = *(const v8bf*)(p);
    v8bf c1 = *(const v8bf*)(p + 16);
    return cat16(c0, c1);
}

// B 32x16 (KxN): lanes 0-15 hold col N=lane, K = {0..7, 16..23}; lanes 16-31 hold
// col N=lane-16, K = {8..15, 24..31}.  Two contiguous 8-elem (16B) chunks.
__device__ __forceinline__ v16bf load_b_frag(const bf16_t* __restrict__ W, int ldw,
                                             int col, int k0, int lane) {
    const bf16_t* p = W + (size_t)col * ldw + k0 + ((lane >> 4) << 3);
    v8bf c0 = *(const v8bf*)(p);
    v8bf c1 = *(const v8bf*)(p + 16);
    return cat16(c0, c1);
}

// ---------------- generic bf16 GEMM:  C[M,Nout] = A[M,K] * W[Nout,K]^T --------
// block = (32,4): 4 waves; each wave computes a 64(M) x 16(N) strip.
// Software-pipelined: fragments for k+32 are in flight while the current
// tile's WMMAs issue, so s_wait_loadcnt retires against older loads instead
// of stalling every matrix op.  M, Nout, K all multiples of 32 here (K >= 64).
__global__ void gemm_bf16_k(const bf16_t* __restrict__ A, const bf16_t* __restrict__ W,
                            float* __restrict__ C, int K, int Nout) {
    const int lane = threadIdx.x;
    const int n0   = blockIdx.x * 16;
    const int m0   = (blockIdx.y * 4 + threadIdx.y) * 64;
    const int ar   = lane & 15;

    v8f acc[4] = {};
    v16bf bcur = load_b_frag(W, K, n0 + ar, 0, lane);
    v16bf acur[4];
#pragma unroll
    for (int t = 0; t < 4; ++t) acur[t] = load_a_frag(A, K, m0 + t * 16 + ar, 0, lane);

    for (int k0 = 0; k0 < K - 32; k0 += 32) {
        const int kn = k0 + 32;
        // near-scope prefetch of the A stream four k-tiles (128 elems) ahead
        __builtin_prefetch(A + (size_t)(m0 + ar) * K + k0 + 128, 0, 3);
        // issue next-tile loads first (independent registers)
        v16bf bn = load_b_frag(W, K, n0 + ar, kn, lane);
        v16bf an[4];
#pragma unroll
        for (int t = 0; t < 4; ++t) an[t] = load_a_frag(A, K, m0 + t * 16 + ar, kn, lane);
        // matrix ops on the already-resident fragments
#pragma unroll
        for (int t = 0; t < 4; ++t)
            acc[t] = __builtin_amdgcn_wmma_f32_16x16x32_bf16(
                false, acur[t], false, bcur, (short)0, acc[t], false, false);
        bcur = bn;
#pragma unroll
        for (int t = 0; t < 4; ++t) acur[t] = an[t];
    }
#pragma unroll
    for (int t = 0; t < 4; ++t)
        acc[t] = __builtin_amdgcn_wmma_f32_16x16x32_bf16(
            false, acur[t], false, bcur, (short)0, acc[t], false, false);

    // D/C layout: VGPR v -> M = v (+8 for upper half-wave), N = lane&15
    const int mb = (lane >> 4) << 3;
#pragma unroll
    for (int t = 0; t < 4; ++t)
#pragma unroll
        for (int v = 0; v < 8; ++v)
            C[(size_t)(m0 + t * 16 + mb + v) * Nout + (n0 + ar)] = acc[t][v];
}

// ---------------- x @ E_w^T with fused fp32->bf16 convert, K tail guard -------
__device__ __forceinline__ v8bf cvt8_guard(const float* __restrict__ row, int k, int K) {
    v8bf r;
    if (k + 8 <= K) {
#pragma unroll
        for (int i = 0; i < 8; ++i) r[i] = (__bf16)row[k + i];
    } else {
#pragma unroll
        for (int i = 0; i < 8; ++i) r[i] = (k + i < K) ? (__bf16)row[k + i] : (__bf16)0.0f;
    }
    return r;
}

__global__ void gemm_xe_k(const float* __restrict__ A, const float* __restrict__ W,
                          float* __restrict__ C, int K, int Kpad, int Nout) {
    const int lane = threadIdx.x;
    const int n0   = blockIdx.x * 16;
    const int m0   = (blockIdx.y * 4 + threadIdx.y) * 64;
    const int ar   = lane & 15;

    v8f acc[4] = {};
    for (int k0 = 0; k0 < Kpad; k0 += 32) {
        const int bk = k0 + ((lane >> 4) << 3);
        const float* wrow = W + (size_t)(n0 + ar) * K;
        v16bf b = cat16(cvt8_guard(wrow, bk, K), cvt8_guard(wrow, bk + 16, K));
        const int ak = k0 + ((lane >> 4) << 3);
#pragma unroll
        for (int t = 0; t < 4; ++t) {
            const float* arow = A + (size_t)(m0 + t * 16 + ar) * K;
            v16bf a = cat16(cvt8_guard(arow, ak, K), cvt8_guard(arow, ak + 16, K));
            acc[t] = __builtin_amdgcn_wmma_f32_16x16x32_bf16(
                false, a, false, b, (short)0, acc[t], false, false);
        }
    }
    const int mb = (lane >> 4) << 3;
#pragma unroll
    for (int t = 0; t < 4; ++t)
#pragma unroll
        for (int v = 0; v < 8; ++v)
            C[(size_t)(m0 + t * 16 + mb + v) * Nout + (n0 + ar)] = acc[t][v];
}

// ---------------- elementwise / utility kernels ----------------
__global__ void convert_bf16_k(const float* __restrict__ src, bf16_t* __restrict__ dst,
                               size_t n) {
    size_t i = (size_t)blockIdx.x * blockDim.x + threadIdx.x;
    if (i < n) dst[i] = (bf16_t)src[i];
}

__global__ void fill0_k(uint32_t* __restrict__ p, size_t n) {
    size_t i = (size_t)blockIdx.x * blockDim.x + threadIdx.x;
    if (i < n) p[i] = 0u;
}

// hs = mask * (adj@h); writes masked fp32 in place + bf16 copy for gate GEMMs
__global__ void mask_k(float* __restrict__ hs, const int* __restrict__ level, int j,
                       bf16_t* __restrict__ hsb) {
    size_t i = (size_t)blockIdx.x * blockDim.x + threadIdx.x;
    if (i >= (size_t)NN * HH) return;
    int row = (int)(i >> 9);                     // /HH
    float v = (level[row] == j) ? hs[i] : 0.0f;
    hs[i]  = v;
    hsb[i] = (bf16_t)v;
}

// r/z gates; stash z (fp32) and (hs*r) in bf16 for the Uh GEMM
__global__ void rz_k(const float* __restrict__ xr, const float* __restrict__ xz,
                     const float* __restrict__ ur, const float* __restrict__ uz,
                     const float* __restrict__ hs, float* __restrict__ zb,
                     bf16_t* __restrict__ hsrb) {
    size_t i = (size_t)blockIdx.x * blockDim.x + threadIdx.x;
    if (i >= (size_t)NN * HH) return;
    float r = 1.0f / (1.0f + __expf(-(xr[i] + ur[i])));
    float z = 1.0f / (1.0f + __expf(-(xz[i] + uz[i])));
    zb[i]   = z;
    hsrb[i] = (bf16_t)(hs[i] * r);
}

// hj = (1-z)*hs + z*tanh(xh+uh); h += mask*hj; refresh bf16 h^T for next adj@h
__global__ void update_k(const float* __restrict__ xh, const float* __restrict__ uh,
                         const float* __restrict__ zb, const float* __restrict__ hs,
                         const int* __restrict__ level, int j,
                         float* __restrict__ h, bf16_t* __restrict__ hTb) {
    size_t i = (size_t)blockIdx.x * blockDim.x + threadIdx.x;
    if (i >= (size_t)NN * HH) return;
    int row = (int)(i >> 9);
    int col = (int)(i & (HH - 1));
    float z  = zb[i];
    float s  = hs[i];
    float hj = (1.0f - z) * s + z * tanhf(xh[i] + uh[i]);
    float hv = h[i];
    if (level[row] == j) { hv += hj; h[i] = hv; }
    hTb[(size_t)col * NN + row] = (bf16_t)hv;   // transposed bf16 copy
}

__global__ void decoder_k(const float* __restrict__ h, const float* __restrict__ dw,
                          const float* __restrict__ db, float* __restrict__ out) {
    size_t i = (size_t)blockIdx.x * blockDim.x + threadIdx.x;
    if (i >= (size_t)NN * CC) return;
    int row = (int)(i >> 2);
    int c   = (int)(i & 3);
    float s = db[c];
    const float* hr = h + (size_t)row * HH;
    const float* wr = dw + (size_t)c * HH;
#pragma unroll 4
    for (int k = 0; k < HH; ++k) s += hr[k] * wr[k];
    out[i] = s;
}

// ---------------- host orchestration ----------------
extern "C" void kernel_launch(void* const* d_in, const int* in_sizes, int n_in,
                              void* d_out, int out_size, void* d_ws, size_t ws_size,
                              hipStream_t stream) {
    (void)in_sizes; (void)n_in; (void)out_size; (void)ws_size;
    const float* x     = (const float*)d_in[0];
    const float* adj   = (const float*)d_in[1];
    const int*   level = (const int*)  d_in[2];
    const float* E_w   = (const float*)d_in[3];
    const float* Wr    = (const float*)d_in[4];
    const float* Wz    = (const float*)d_in[5];
    const float* Ur    = (const float*)d_in[6];
    const float* Uz    = (const float*)d_in[7];
    const float* Wh    = (const float*)d_in[8];
    const float* Uh    = (const float*)d_in[9];
    const float* dec_w = (const float*)d_in[10];
    const float* dec_b = (const float*)d_in[11];
    float* out = (float*)d_out;

    // carve workspace
    char* ws = (char*)d_ws;
    size_t off = 0;
    auto carve = [&](size_t bytes) -> void* {
        void* p = ws + off;
        off = (off + bytes + 255) & ~(size_t)255;
        return p;
    };
    const size_t NH = (size_t)NN * HH;
    bf16_t* adjb = (bf16_t*)carve((size_t)NN * NN * 2);
    bf16_t* Wrb  = (bf16_t*)carve((size_t)HH * HH * 2);
    bf16_t* Wzb  = (bf16_t*)carve((size_t)HH * HH * 2);
    bf16_t* Whb  = (bf16_t*)carve((size_t)HH * HH * 2);
    bf16_t* Urb  = (bf16_t*)carve((size_t)HH * HH * 2);
    bf16_t* Uzb  = (bf16_t*)carve((size_t)HH * HH * 2);
    bf16_t* Uhb  = (bf16_t*)carve((size_t)HH * HH * 2);
    float*  xhat = (float*) carve(NH * 4);
    bf16_t* xhb  = (bf16_t*)carve(NH * 2);
    float*  xr   = (float*) carve(NH * 4);
    float*  xz   = (float*) carve(NH * 4);
    float*  xh   = (float*) carve(NH * 4);
    float*  h    = (float*) carve(NH * 4);
    bf16_t* hTb  = (bf16_t*)carve(NH * 2);
    float*  hs   = (float*) carve(NH * 4);   // also holds raw adj@h
    bf16_t* hsb  = (bf16_t*)carve(NH * 2);
    float*  ur   = (float*) carve(NH * 4);
    float*  uz   = (float*) carve(NH * 4);
    float*  uhb  = (float*) carve(NH * 4);
    float*  zb   = (float*) carve(NH * 4);
    bf16_t* hsrb = (bf16_t*)carve(NH * 2);

    const dim3 gblk(32, 4);                    // 4 waves (wave32)
    const dim3 ggrid(HH / 16, NN / 256);       // every GEMM here is 4096 x 512
    const int EW = 256;
    const int nbNH = (int)((NH + EW - 1) / EW);

    // one-time bf16 conversions (adj stays L2-resident at 33.5 MB thereafter)
    convert_bf16_k<<<(int)(((size_t)NN * NN + EW - 1) / EW), EW, 0, stream>>>(adj, adjb, (size_t)NN * NN);
    const size_t HH2 = (size_t)HH * HH;
    convert_bf16_k<<<(int)((HH2 + EW - 1) / EW), EW, 0, stream>>>(Wr, Wrb, HH2);
    convert_bf16_k<<<(int)((HH2 + EW - 1) / EW), EW, 0, stream>>>(Wz, Wzb, HH2);
    convert_bf16_k<<<(int)((HH2 + EW - 1) / EW), EW, 0, stream>>>(Wh, Whb, HH2);
    convert_bf16_k<<<(int)((HH2 + EW - 1) / EW), EW, 0, stream>>>(Ur, Urb, HH2);
    convert_bf16_k<<<(int)((HH2 + EW - 1) / EW), EW, 0, stream>>>(Uz, Uzb, HH2);
    convert_bf16_k<<<(int)((HH2 + EW - 1) / EW), EW, 0, stream>>>(Uh, Uhb, HH2);

    // x_hat = x @ E_w^T   (K=5000 padded to 5024 in-register)
    gemm_xe_k<<<ggrid, gblk, 0, stream>>>(x, E_w, xhat, DD, 5024, HH);
    convert_bf16_k<<<nbNH, EW, 0, stream>>>(xhat, xhb, NH);

    // gate input projections (constant across levels)
    gemm_bf16_k<<<ggrid, gblk, 0, stream>>>(xhb, Wrb, xr, HH, HH);
    gemm_bf16_k<<<ggrid, gblk, 0, stream>>>(xhb, Wzb, xz, HH, HH);
    gemm_bf16_k<<<ggrid, gblk, 0, stream>>>(xhb, Whb, xh, HH, HH);

    // h = 0, hT(bf16) = 0
    fill0_k<<<nbNH, EW, 0, stream>>>((uint32_t*)h, NH);
    fill0_k<<<(int)((NH / 2 + EW - 1) / EW), EW, 0, stream>>>((uint32_t*)hTb, NH / 2);

    // bottom-up tree GRU
    for (int j = LL - 1; j >= 0; --j) {
        gemm_bf16_k<<<ggrid, gblk, 0, stream>>>(adjb, hTb, hs, NN, HH);        // adj @ h
        mask_k<<<nbNH, EW, 0, stream>>>(hs, level, j, hsb);
        gemm_bf16_k<<<ggrid, gblk, 0, stream>>>(hsb, Urb, ur, HH, HH);         // hs @ Ur^T
        gemm_bf16_k<<<ggrid, gblk, 0, stream>>>(hsb, Uzb, uz, HH, HH);         // hs @ Uz^T
        rz_k<<<nbNH, EW, 0, stream>>>(xr, xz, ur, uz, hs, zb, hsrb);
        gemm_bf16_k<<<ggrid, gblk, 0, stream>>>(hsrb, Uhb, uhb, HH, HH);       // (hs*r) @ Uh^T
        update_k<<<nbNH, EW, 0, stream>>>(xh, uhb, zb, hs, level, j, h, hTb);
    }

    // decoder
    decoder_k<<<(int)(((size_t)NN * CC + EW - 1) / EW), EW, 0, stream>>>(h, dec_w, dec_b, out);
}
